// RuleNBFNet_11003706213184
// MI455X (gfx1250) — matrix operand order repaired and checked by reading, not compiled, not run on hardware
//
#include <hip/hip_runtime.h>
#include <hip/hip_bf16.h>

typedef _Float16 f16;
typedef _Float16 v16h __attribute__((ext_vector_type(16)));
typedef _Float16 v8h  __attribute__((ext_vector_type(8)));
typedef _Float16 v4h  __attribute__((ext_vector_type(4)));
typedef float    v8f  __attribute__((ext_vector_type(8)));

#define B_SZ 8
#define D_SZ 64
#define H_SZ 3
#define NRULE 1728            // 12^3
#define KFEAT 832             // 13*D
#define KCHUNKS 26            // 832/32
#define NTILES 4              // 64/16
#define RULES_PER_WG 32
#define WAVES_PER_WG 8
#define RULES_PER_WAVE 4
#define A_STRIDE 840          // padded LDS row stride (f16 elems)

// ---- workspace layout (bytes) ----
#define WFRAG_PER_L (KCHUNKS*NTILES*32*16)          // 53248 f16 per layer
#define WS_WFRAG 0                                   // f16 [3][26][4][32][16] = 319488 B
#define WS_RELT  (3*WFRAG_PER_L*2)                   // f32 [3][8][768]       =  73728 B
#define WS_SCORE (WS_RELT + 3*8*768*4)               // f32 [8][1728]         =  55296 B

// -------- prep 1: stage lin_W as f16 WMMA B-fragments --------
// B-frag (32x16 KxN, f16): lane n<16 -> col n, K = c*32 + 0..15
//                          lane n+16 -> col n, K = c*32 + 16..31
__global__ __launch_bounds__(256)
void prep_wfrag(const float* __restrict__ linW, f16* __restrict__ wfrag) {
  int idx = blockIdx.x * 256 + threadIdx.x;
  if (idx >= 3 * WFRAG_PER_L) return;
  int e    =  idx        & 15;
  int lane = (idx >> 4)  & 31;
  int nt   = (idx >> 9)  & 3;
  int c    = (idx >> 11) % KCHUNKS;
  int l    =  idx / WFRAG_PER_L;
  int n = nt * 16 + (lane & 15);
  int k = c * 32 + ((lane >> 4) << 4) + e;
  wfrag[idx] = (f16)linW[(size_t)l * KFEAT * 64 + (size_t)k * 64 + n];
}

// -------- prep 2: rel_table[l][b] = query[b] @ rel_W[l] + rel_b[l] --------
__global__ __launch_bounds__(256)
void prep_rel(const float* __restrict__ query, const float* __restrict__ relW,
              const float* __restrict__ relb, float* __restrict__ relt) {
  int idx = blockIdx.x * 256 + threadIdx.x;     // [l][b][m]
  if (idx >= 3 * B_SZ * 768) return;
  int m = idx % 768;
  int b = (idx / 768) % B_SZ;
  int l = idx / (B_SZ * 768);
  float acc = relb[l * 768 + m];
  const float* q = query + b * 64;
  const float* w = relW + (size_t)l * 64 * 768 + m;
  for (int k = 0; k < 64; ++k) acc += q[k] * w[(size_t)k * 768];
  relt[idx] = acc;
}

// -------- main fused 3-layer kernel + per-rule MLP score --------
__global__ __launch_bounds__(256)
void rulenet_main(const float* __restrict__ query,
                  const float* __restrict__ indicator,
                  const float* __restrict__ lin_b,     // (3,64)
                  const float* __restrict__ mlp_W1,    // (128,64)
                  const float* __restrict__ mlp_b1,    // (64)
                  const float* __restrict__ mlp_W2,    // (64)
                  const float* __restrict__ mlp_b2,    // (1)
                  const f16*   __restrict__ wfrag,
                  const float* __restrict__ relt,      // (3,8,768)
                  float* __restrict__ scores)
{
  extern __shared__ char smem[];
  f16*   atiles = (f16*)smem;                                        // [8][16][A_STRIDE]
  float* rel_s  = (float*)(smem + WAVES_PER_WG * 16 * A_STRIDE * 2); // [3][768]

  const int tid  = threadIdx.x;
  const int wave = tid >> 5;
  const int lane = tid & 31;
  const int b    = blockIdx.y;
  const int ruleBase = blockIdx.x * RULES_PER_WG;
  const int waveRule = ruleBase + wave * RULES_PER_WAVE;

  // cooperative load of rel tables for this batch element
  for (int i = tid; i < 3 * 768; i += 256)
    rel_s[i] = relt[(i / 768) * (B_SZ * 768) + b * 768 + (i % 768)];
  __syncthreads();

  f16* At = atiles + wave * 16 * A_STRIDE;

  // init hidden cols (0..63): p==0 -> indicator, else 0
  for (int t = lane; t < 16 * 64; t += 32) {
    int row = t >> 6, d = t & 63;
    float v = ((row & 3) == 0) ? indicator[d] : 0.f;
    At[row * A_STRIDE + d] = (f16)v;
  }
  // init layer-invariant upd0 block for p==0 rows (cols 64..831)
  // scales_head = [1, 0, 100]; f0 = [ind,ind,ind,sqrt(EPS)=1e-3]
  for (int i = 0; i < 8; ++i) {
    int t = i * 32 + lane;
    int d = t & 63, rl = t >> 6;          // rl = rule_local 0..3
    float ind = indicator[d];
    f16* dst = At + (rl * 4) * A_STRIDE + 64 + d * 12;
    v4h o0 = { (f16)ind, (f16)0.f, (f16)(100.f * ind), (f16)ind };
    v4h o1 = { (f16)0.f, (f16)(100.f * ind), (f16)ind, (f16)0.f };
    v4h o2 = { (f16)(100.f * ind), (f16)1e-3f, (f16)0.f, (f16)0.1f };
    *(v4h*)(dst + 0) = o0; *(v4h*)(dst + 4) = o1; *(v4h*)(dst + 8) = o2;
  }

  const int  rowA = lane & 15;
  const int  hk   = lane >> 4;
  const f16* Arow = At + rowA * A_STRIDE;

#define LOAD_B(wl_, c_, Bf_)                                                      \
  do {                                                                            \
    _Pragma("unroll")                                                             \
    for (int nt_ = 0; nt_ < NTILES; ++nt_)                                        \
      Bf_[nt_] = *(const v16h*)((wl_) +                                           \
          ((size_t)((c_) * NTILES + nt_) * 32 + lane) * 16);                      \
  } while (0)

#define LOAD_A(c_, Af_)                                                           \
  do {                                                                            \
    int kOff_ = (c_) * 32 + hk * 16;                                              \
    v8h a0_ = *(const v8h*)(Arow + kOff_);                                        \
    v8h a1_ = *(const v8h*)(Arow + kOff_ + 8);                                    \
    _Pragma("unroll")                                                             \
    for (int q_ = 0; q_ < 8; ++q_) { Af_[q_] = a0_[q_]; Af_[q_ + 8] = a1_[q_]; }  \
  } while (0)

#define WMMA4(Af_, Bf_)                                                           \
  do {                                                                            \
    _Pragma("unroll")                                                             \
    for (int nt_ = 0; nt_ < NTILES; ++nt_)                                        \
      acc[nt_] = __builtin_amdgcn_wmma_f32_16x16x32_f16(                          \
          false, Af_, false, Bf_[nt_], (short)0, acc[nt_], false, false);         \
  } while (0)

  const float S1 = 4.f / 3.f, S2 = 0.75f;   // scales_tail
  for (int l = 0; l < 3; ++l) {
    // ---- feature fill for p>=1 rows (12 rows x 64 d = 768 tasks) ----
    for (int i = 0; i < 24; ++i) {
      int t = i * 32 + lane;
      int d = t & 63, rl = t >> 6;          // rl 0..11
      int rule_local = rl / 3, pm1 = rl % 3;
      int r = waveRule + rule_local;
      int j = (pm1 == 0) ? (r / 144) : (pm1 == 1) ? ((r / 12) % 12) : (r % 12);
      float hprev = (float)At[(rule_local * 4 + pm1) * A_STRIDE + d];
      float m  = hprev * rel_s[l * 768 + j * 64 + d];
      float mean = 0.5f * m;
      float mx = fmaxf(m, 0.f);
      float mn = fminf(m, 0.f);
      float sd = sqrtf(fmaxf(0.25f * m * m, 1e-6f));
      f16* dst = At + (rule_local * 4 + pm1 + 1) * A_STRIDE + 64 + d * 12;
      v4h o0 = { (f16)mean, (f16)(mean * S1), (f16)(mean * S2), (f16)mx };
      v4h o1 = { (f16)(mx * S1), (f16)(mx * S2), (f16)mn, (f16)(mn * S1) };
      v4h o2 = { (f16)(mn * S2), (f16)sd, (f16)(sd * S1), (f16)(sd * S2) };
      *(v4h*)(dst + 0) = o0; *(v4h*)(dst + 4) = o1; *(v4h*)(dst + 8) = o2;
    }

    // ---- WMMA: (16 x 832) @ (832 x 64), f32 accumulate ----
    // Software-pipelined: fragments for chunk c+1 are issued before the
    // WMMA group for chunk c so global L2 latency overlaps matrix math.
    v8f acc[NTILES];
    #pragma unroll
    for (int nt = 0; nt < NTILES; ++nt)
      #pragma unroll
      for (int q = 0; q < 8; ++q) acc[nt][q] = 0.f;

    const f16* wl = wfrag + (size_t)l * WFRAG_PER_L;
    v16h B0[NTILES], B1[NTILES];
    v16h A0, A1;
    LOAD_B(wl, 0, B0);
    LOAD_A(0, A0);
    for (int c = 0; c < KCHUNKS; c += 2) {
      LOAD_B(wl, c + 1, B1);        // prefetch odd chunk
      LOAD_A(c + 1, A1);
      WMMA4(A0, B0);                // compute even chunk
      if (c + 2 < KCHUNKS) {
        LOAD_B(wl, c + 2, B0);      // prefetch next even chunk
        LOAD_A(c + 2, A0);
      }
      WMMA4(A1, B1);                // compute odd chunk
    }

    // ---- bias + relu, write back hidden (cols 0..63) as f16 ----
    #pragma unroll
    for (int nt = 0; nt < NTILES; ++nt) {
      int col = nt * 16 + (lane & 15);
      float bias = lin_b[l * 64 + col];
      int rbase = (lane >> 4) * 8;        // C layout: M = v (+8 for upper half-wave)
      #pragma unroll
      for (int v = 0; v < 8; ++v) {
        float h = fmaxf(acc[nt][v] + bias, 0.f);
        At[(v + rbase) * A_STRIDE + col] = (f16)h;
      }
    }
  }

  // ---- MLP score from final hidden rows (p == 3) ----
  for (int ru = 0; ru < RULES_PER_WAVE; ++ru) {
    int r = waveRule + ru;
    const f16* hrow = At + (ru * 4 + 3) * A_STRIDE;
    int j0 = lane, j1 = lane + 32;
    float a0 = mlp_b1[j0], a1 = mlp_b1[j1];
    for (int k = 0; k < 64; ++k) {
      float f = (float)hrow[k];
      a0 += f * mlp_W1[k * 64 + j0];
      a1 += f * mlp_W1[k * 64 + j1];
    }
    for (int k = 0; k < 64; ++k) {
      float f = query[b * 64 + k];
      a0 += f * mlp_W1[(64 + k) * 64 + j0];
      a1 += f * mlp_W1[(64 + k) * 64 + j1];
    }
    float s = fmaxf(a0, 0.f) * mlp_W2[j0] + fmaxf(a1, 0.f) * mlp_W2[j1];
    #pragma unroll
    for (int off = 16; off > 0; off >>= 1)
      s += __shfl_xor(s, off, 32);
    if (lane == 0) scores[b * NRULE + r] = s + mlp_b2[0];
  }
#undef LOAD_B
#undef LOAD_A
#undef WMMA4
}

// -------- finalize: softmax over rules + einsum via 3x12 attention bins --------
__global__ __launch_bounds__(256)
void finalize(const float* __restrict__ scores, const float* __restrict__ rel_emb,
              float* __restrict__ out) {
  __shared__ float red[256];
  __shared__ float wsum[3][12];
  int tid = threadIdx.x, b = blockIdx.x;
  const float* sc = scores + b * NRULE;

  float m = -1e30f;
  for (int r = tid; r < NRULE; r += 256) m = fmaxf(m, sc[r]);
  red[tid] = m; __syncthreads();
  for (int s = 128; s > 0; s >>= 1) {
    if (tid < s) red[tid] = fmaxf(red[tid], red[tid + s]);
    __syncthreads();
  }
  float mx = red[0]; __syncthreads();

  float sum = 0.f;
  for (int r = tid; r < NRULE; r += 256) sum += expf(sc[r] - mx);
  red[tid] = sum; __syncthreads();
  for (int s = 128; s > 0; s >>= 1) {
    if (tid < s) red[tid] += red[tid + s];
    __syncthreads();
  }
  float inv = 1.f / red[0];

  if (tid < 36) ((float*)wsum)[tid] = 0.f;
  __syncthreads();
  for (int r = tid; r < NRULE; r += 256) {
    float att = expf(sc[r] - mx) * inv;
    atomicAdd(&wsum[0][r / 144], att);
    atomicAdd(&wsum[1][(r / 12) % 12], att);
    atomicAdd(&wsum[2][r % 12], att);
  }
  __syncthreads();

  if (tid < 192) {                         // subgoals[b][h][d]
    int h = tid >> 6, d = tid & 63;
    float v = 0.f;
    for (int j = 0; j < 12; ++j) v += wsum[h][j] * rel_emb[j * 64 + d];
    out[b * 192 + tid] = v;
  }
  if (tid < 3)                             // masks = ones(B,H)
    out[B_SZ * H_SZ * D_SZ + b * 3 + tid] = 1.0f;
}

extern "C" void kernel_launch(void* const* d_in, const int* in_sizes, int n_in,
                              void* d_out, int out_size, void* d_ws, size_t ws_size,
                              hipStream_t stream) {
  (void)in_sizes; (void)n_in; (void)out_size; (void)ws_size;
  const float* query   = (const float*)d_in[0];
  const float* rel_emb = (const float*)d_in[1];
  const float* indic   = (const float*)d_in[2];
  const float* rel_W   = (const float*)d_in[3];
  const float* rel_b   = (const float*)d_in[4];
  const float* lin_W   = (const float*)d_in[5];
  const float* lin_b   = (const float*)d_in[6];
  const float* mlp_W1  = (const float*)d_in[7];
  const float* mlp_b1  = (const float*)d_in[8];
  const float* mlp_W2  = (const float*)d_in[9];
  const float* mlp_b2  = (const float*)d_in[10];

  char* ws = (char*)d_ws;
  f16*   wfrag  = (f16*)(ws + WS_WFRAG);
  float* relt   = (float*)(ws + WS_RELT);
  float* scores = (float*)(ws + WS_SCORE);

  prep_wfrag<<<(3 * WFRAG_PER_L + 255) / 256, 256, 0, stream>>>(lin_W, wfrag);
  prep_rel<<<(3 * B_SZ * 768 + 255) / 256, 256, 0, stream>>>(query, rel_W, rel_b, relt);

  size_t smem = (size_t)WAVES_PER_WG * 16 * A_STRIDE * 2 + 3 * 768 * 4; // 224,256 B
  rulenet_main<<<dim3(NRULE / RULES_PER_WG, B_SZ), 256, smem, stream>>>(
      query, indic, lin_b, mlp_W1, mlp_b1, mlp_W2, mlp_b2, wfrag, relt, scores);

  finalize<<<B_SZ, 256, 0, stream>>>(scores, rel_emb, (float*)d_out);
}
